// SoftAttnGRU_82197084110889
// MI455X (gfx1250) — matrix operand order repaired and checked by reading, not compiled, not run on hardware
//
#include <hip/hip_runtime.h>
#include <hip/hip_bf16.h>
#include <cstdint>

#define UNITS  1024
#define BATCH  128
#define TSTEPS 256
#define XROW   (UNITS + 1)   // 1025 (last col = attention weight)
#define G3     (3 * UNITS)   // 3072

#define KCHUNK      128                        // K elements staged per pipeline stage
#define NCOLS       16                         // N columns per block
#define SLABS       6                          // {z,r,h} x {kernel, recurrent}
#define COLBYTES    (KCHUNK * 2)               // 256 B per (column, chunk)
#define CHUNK_BYTES (SLABS * NCOLS * COLBYTES) // 24576 B
#define UNITS16     (SLABS * NCOLS * KCHUNK * 2 / 16) // 1536 16-byte units per chunk

typedef __attribute__((ext_vector_type(16))) __bf16 v16bf;
typedef __attribute__((ext_vector_type(8)))  __bf16 v8bf;
typedef __attribute__((ext_vector_type(8)))  float  v8f;

__device__ __forceinline__ unsigned short f32_to_bf16_rne(float f) {
    union { float f; uint32_t u; } c; c.f = f;
    uint32_t u = c.u;
    uint32_t r = u + 0x7FFFu + ((u >> 16) & 1u);   // round-to-nearest-even
    return (unsigned short)(r >> 16);
}

// ---------------- Phase 0: weight transpose + bf16 convert ----------------
// src: (U, 3U) row-major f32   dst: (3U, U) row-major bf16  (dst[n*U+k] = src[k*3U+n])
__global__ void transpose_w_bf16(const float* __restrict__ src,
                                 unsigned short* __restrict__ dst) {
    int idx = blockIdx.x * blockDim.x + threadIdx.x;
    if (idx >= G3 * UNITS) return;
    int n = idx / UNITS;
    int k = idx - n * UNITS;
    dst[idx] = f32_to_bf16_rne(src[(size_t)k * G3 + n]);
}

// x: (B, T, 1025) f32 -> xb: (T, B, U) bf16 time-major, attn: (T, B) f32
__global__ void convert_x_bf16(const float* __restrict__ x,
                               unsigned short* __restrict__ xb,
                               float* __restrict__ attn) {
    size_t idx = (size_t)blockIdx.x * blockDim.x + threadIdx.x;
    size_t total = (size_t)TSTEPS * BATCH * UNITS;
    if (idx >= total) return;
    int u = (int)(idx % UNITS);
    int b = (int)((idx / UNITS) % BATCH);
    int t = (int)(idx / ((size_t)UNITS * BATCH));
    const float* row = x + ((size_t)b * TSTEPS + t) * XROW;
    xb[idx] = f32_to_bf16_rne(row[u]);
    if (u == 0) attn[(size_t)t * BATCH + b] = row[UNITS];
}

__global__ void init_h(float* __restrict__ hf, unsigned short* __restrict__ hb) {
    int idx = blockIdx.x * blockDim.x + threadIdx.x;
    if (idx >= BATCH * UNITS) return;
    hf[idx] = 0.0f;
    hb[idx] = 0;      // bf16 zero
}

// ---------------- helpers ----------------
__device__ __forceinline__ v16bf load_afrag(const unsigned short* p) {
    // A frag (this code's order, same as B): lane's 16 elements are 16 contiguous k
    v8bf lo = *(const v8bf*)(p);
    v8bf hi = *(const v8bf*)(p + 8);
    return __builtin_shufflevector(lo, hi, 0,1,2,3,4,5,6,7,8,9,10,11,12,13,14,15);
}

__device__ __forceinline__ v16bf lds_bfrag(const char* buf, int slab, int l15, int koff) {
    // LDS layout: [slab*16 + col][KCHUNK] bf16, contiguous in k
    return *(const v16bf*)(buf + ((slab * NCOLS + l15) * COLBYTES) + koff * 2);
}

// One complete fragment set for a 32-wide K slice: 2 A-frags + 6 B-frags.
struct Frags {
    v16bf ax, ah, bz, br, bh, cz, cr, ch;
};

__device__ __forceinline__ Frags load_frags(const unsigned short* ax,
                                            const unsigned short* ah,
                                            const char* buf,
                                            int l15, int half, int k0, int kin) {
    Frags f;
    const int koff = kin + 16 * half;       // B-frag k offset within chunk
    f.ax = load_afrag(ax + k0 + kin);
    f.ah = load_afrag(ah + k0 + kin);
    f.bz = lds_bfrag(buf, 0, l15, koff);
    f.br = lds_bfrag(buf, 1, l15, koff);
    f.bh = lds_bfrag(buf, 2, l15, koff);
    f.cz = lds_bfrag(buf, 3, l15, koff);
    f.cr = lds_bfrag(buf, 4, l15, koff);
    f.ch = lds_bfrag(buf, 5, l15, koff);
    return f;
}

__device__ __forceinline__ void do_wmmas(const Frags& f, v8f& accz, v8f& accr,
                                         v8f& accxh, v8f& acchr) {
    accz  = __builtin_amdgcn_wmma_f32_16x16x32_bf16(false, f.ax, false, f.bz, (short)0, accz,  false, false);
    accr  = __builtin_amdgcn_wmma_f32_16x16x32_bf16(false, f.ax, false, f.br, (short)0, accr,  false, false);
    accxh = __builtin_amdgcn_wmma_f32_16x16x32_bf16(false, f.ax, false, f.bh, (short)0, accxh, false, false);
    accz  = __builtin_amdgcn_wmma_f32_16x16x32_bf16(false, f.ah, false, f.cz, (short)0, accz,  false, false);
    accr  = __builtin_amdgcn_wmma_f32_16x16x32_bf16(false, f.ah, false, f.cr, (short)0, accr,  false, false);
    acchr = __builtin_amdgcn_wmma_f32_16x16x32_bf16(false, f.ah, false, f.ch, (short)0, acchr, false, false);
}

// IGroupLP schedule pin for one 32-wide K slice:
//   12 DS reads (6 B-frags) -> 4 VMEM reads (2 A-frags) -> 6 WMMAs.
// Forces batched loads into disjoint registers (one s_wait per 6 WMMAs)
// instead of the scheduler's load->wait->wmma singles.
__device__ __forceinline__ void sched_slice() {
    __builtin_amdgcn_sched_group_barrier(0x100, 12, 0); // DS_READ
    __builtin_amdgcn_sched_group_barrier(0x020, 4, 0);  // VMEM_READ
    __builtin_amdgcn_sched_group_barrier(0x008, 6, 0);  // MFMA/WMMA
}

// Cooperative async stage of one K-chunk of all 6 weight slabs into LDS.
// 1536 16-byte units, 256 threads -> 6 async b128 issues per thread (ASYNCcnt += 6).
__device__ __forceinline__ void stage_chunk(const unsigned short* __restrict__ Kt,
                                            const unsigned short* __restrict__ Rt,
                                            int ncol0, int k0,
                                            unsigned lds_buf_base, int tid) {
    #pragma unroll
    for (int r = 0; r < UNITS16 / 256; ++r) {
        const int u       = tid + r * 256;   // 0..1535
        const int col_lin = u >> 4;          // 0..95
        const int sub     = u & 15;          // 16B unit within 256B column chunk
        const int slab    = col_lin >> 4;    // 0..5
        const int c       = col_lin & 15;
        const unsigned short* W = (slab < 3) ? Kt : Rt;
        const int g = (slab < 3) ? slab : slab - 3;
        const unsigned short* gptr =
            W + ((size_t)(g * UNITS + ncol0 + c) * UNITS + k0) + sub * 8;
        const unsigned ldsa = lds_buf_base + (unsigned)(u * 16);
        asm volatile("global_load_async_to_lds_b128 %0, %1, off"
                     :: "v"(ldsa), "v"(gptr) : "memory");
    }
}

// ---------------- Per-timestep GRU step (WMMA bf16, async-LDS weight pipeline) ----
// grid = UNITS/16 = 64 blocks (one per 16-col N-chunk), block = 256 (8 waves).
// Wave w owns M-tile w (rows 16w..16w+15); all 8 waves share the staged B slabs.
__global__ __launch_bounds__(256, 1)
void gru_step(const unsigned short* __restrict__ Kt,    // (3U, U) bf16, kernel^T
              const unsigned short* __restrict__ Rt,    // (3U, U) bf16, recurrent^T
              const float*          __restrict__ bias,  // (3U) f32
              const unsigned short* __restrict__ Xb_t,  // (B, U) bf16, x_i at step t
              const float*          __restrict__ attn_t,// (B) f32
              const float*          __restrict__ h_in,  // (B, U) f32
              const unsigned short* __restrict__ hb_in, // (B, U) bf16
              float*                __restrict__ h_out, // (B, U) f32
              unsigned short*       __restrict__ hb_out)// (B, U) bf16
{
    __shared__ char smem[2 * CHUNK_BYTES];   // 48 KB double buffer

    const int tid   = threadIdx.x;
    const int lane  = tid & 31;
    const int wave  = tid >> 5;
    const int half  = lane >> 4;             // K-half select (A & B layouts)
    const int l15   = lane & 15;             // A: row m ; B/C/D: column n
    const int ncol0 = blockIdx.x * NCOLS;    // block's column chunk within U
    const int mbase = wave * 16;             // wave's M-tile (8 waves cover B=128)
    const int mrow  = mbase + l15;
    const int ncol  = ncol0 + l15;

    const unsigned smem_base = (unsigned)(uintptr_t)(void*)smem;

    const unsigned short* ax = Xb_t  + (size_t)mrow * UNITS + 16 * half;
    const unsigned short* ah = hb_in + (size_t)mrow * UNITS + 16 * half;

    // z and r fuse x- and h- contributions; the candidate gate needs
    // x@kh and h@rh kept separate (r multiplies only the recurrent part).
    v8f accz = {}, accr = {}, accxh = {}, acchr = {};

    // Prologue: stage chunk 0.
    stage_chunk(Kt, Rt, ncol0, 0, smem_base, tid);

    const int NCHUNKS = UNITS / KCHUNK;      // 8
    for (int cchunk = 0; cchunk < NCHUNKS; ++cchunk) {
        // Wait for this wave's outstanding async loads (only chunk c in flight),
        // then barrier so every wave sees the fully staged buffer.
        asm volatile("s_wait_asynccnt 0x0" ::: "memory");
        __syncthreads();

        // Kick off the next chunk into the other buffer (overlaps with compute).
        if (cchunk + 1 < NCHUNKS) {
            stage_chunk(Kt, Rt, ncol0, (cchunk + 1) * KCHUNK,
                        smem_base + (unsigned)(((cchunk + 1) & 1) * CHUNK_BYTES), tid);
        }

        const char* buf = smem + (cchunk & 1) * CHUNK_BYTES;
        const int k0 = cchunk * KCHUNK;

        // Software-pipelined inner loop with IGroupLP pins: slice i+1 fragments
        // fetched into disjoint registers while slice i's 6 WMMAs execute.
        Frags cur = load_frags(ax, ah, buf, l15, half, k0, 0);
        #pragma unroll
        for (int kin = 0; kin < KCHUNK - 32; kin += 32) {
            Frags nxt = load_frags(ax, ah, buf, l15, half, k0, kin + 32);
            do_wmmas(cur, accz, accr, accxh, acchr);
            sched_slice();
            cur = nxt;
        }
        do_wmmas(cur, accz, accr, accxh, acchr);
        sched_slice();
        // Restaging over the other buffer is safe: the barrier at the top of
        // the next iteration orders it against all waves' reads.
    }

    // Elementwise gate update. C/D layout: lane -> n = lane&15, VGPR v -> m = v + 8*(lane>>4).
    const float bz_v = bias[ncol];
    const float br_v = bias[UNITS + ncol];
    const float bh_v = bias[2 * UNITS + ncol];
    #pragma unroll
    for (int v = 0; v < 8; ++v) {
        const int m = mbase + v + 8 * half;
        const size_t o = (size_t)m * UNITS + ncol;
        const float ho = h_in[o];
        const float aw = attn_t[m];
        const float z  = fminf(fmaxf(0.2f * (accz[v] + bz_v) + 0.5f, 0.0f), 1.0f);
        const float r  = fminf(fmaxf(0.2f * (accr[v] + br_v) + 0.5f, 0.0f), 1.0f);
        const float hh = tanhf(accxh[v] + bh_v + r * acchr[v]);
        float h = z * ho + (1.0f - z) * hh;
        h = aw * h + (1.0f - aw) * ho;
        h_out[o]  = h;
        hb_out[o] = f32_to_bf16_rne(h);
    }
}

// ---------------- Launcher ----------------
extern "C" void kernel_launch(void* const* d_in, const int* in_sizes, int n_in,
                              void* d_out, int out_size, void* d_ws, size_t ws_size,
                              hipStream_t stream) {
    const float* x    = (const float*)d_in[0];  // (B, T, 1025)
    const float* kern = (const float*)d_in[1];  // (U, 3U)
    const float* rker = (const float*)d_in[2];  // (U, 3U)
    const float* bias = (const float*)d_in[3];  // (3U,)

    char* ws = (char*)d_ws;
    size_t off = 0;
    auto carve = [&](size_t bytes) -> void* {
        void* p = ws + off;
        off += (bytes + 255) & ~(size_t)255;
        return p;
    };
    unsigned short* Kt   = (unsigned short*)carve((size_t)G3 * UNITS * 2);             // 6 MB
    unsigned short* Rt   = (unsigned short*)carve((size_t)G3 * UNITS * 2);             // 6 MB
    unsigned short* Xb   = (unsigned short*)carve((size_t)TSTEPS * BATCH * UNITS * 2); // 64 MB
    float*          Attn = (float*)carve((size_t)TSTEPS * BATCH * 4);
    float*          hf0  = (float*)carve((size_t)BATCH * UNITS * 4);
    float*          hf1  = (float*)carve((size_t)BATCH * UNITS * 4);
    unsigned short* hb0  = (unsigned short*)carve((size_t)BATCH * UNITS * 2);
    unsigned short* hb1  = (unsigned short*)carve((size_t)BATCH * UNITS * 2);
    float*          hf[2] = { hf0, hf1 };
    unsigned short* hb[2] = { hb0, hb1 };

    {
        int n = G3 * UNITS;
        transpose_w_bf16<<<(n + 255) / 256, 256, 0, stream>>>(kern, Kt);
        transpose_w_bf16<<<(n + 255) / 256, 256, 0, stream>>>(rker, Rt);
    }
    {
        size_t n = (size_t)TSTEPS * BATCH * UNITS;
        convert_x_bf16<<<(unsigned)((n + 255) / 256), 256, 0, stream>>>(x, Xb, Attn);
    }
    init_h<<<(BATCH * UNITS + 255) / 256, 256, 0, stream>>>(hf[0], hb[0]);

    dim3 grid(UNITS / NCOLS);   // 64 blocks, 8 waves each; B-slabs shared per block
    for (int t = 0; t < TSTEPS; ++t) {
        const int cur = t & 1, nxt = cur ^ 1;
        float* hout = (t == TSTEPS - 1) ? (float*)d_out : hf[nxt];
        gru_step<<<grid, 256, 0, stream>>>(
            Kt, Rt, bias,
            Xb + (size_t)t * BATCH * UNITS,
            Attn + (size_t)t * BATCH,
            hf[cur], hb[cur],
            hout, hb[nxt]);
    }
}
